// LSTM_6038724019003
// MI455X (gfx1250) — matrix operand order repaired and checked by reading, not compiled, not run on hardware
//
#include <hip/hip_runtime.h>

// ---------------------------------------------------------------------------
// LSTM for MI455X (gfx1250): bf16 WMMA (16x16x32) with fp32 accumulate.
//   T=512, B=64, I=1024, H=1024.
// Phase 1: Xg = X @ [Wa_x|Wi_x|Wf_x|Wo_x] + b  (parallel GEMM, 4 N-tiles/wave,
//          software-pipelined, output in WMMA C-fragment order, bf16)
// Phase 2: 512 serialized step kernels: g = Xg[t] + h @ Wh ; gate math.
//          Each wave: 2 M-tiles x 4 gates (8 WMMA chains, shared B frags),
//          software-pipelined, fast rcp-based sigmoid/tanh.
// Weights repacked once into WMMA B-fragment lane order (bf16).
// ---------------------------------------------------------------------------

typedef __attribute__((ext_vector_type(16))) __bf16 v16bf;
typedef __attribute__((ext_vector_type(8)))  __bf16 v8bf;
typedef __attribute__((ext_vector_type(8)))  float  v8f;
typedef __attribute__((ext_vector_type(4)))  float  v4f;

#define T_DIM 512
#define B_DIM 64
#define I_DIM 1024
#define H_DIM 1024
#define G4H   (4 * H_DIM)

// Packed B-fragment layout for a [1024,1024] weight matrix W (row=k, col=n):
//   packed[gate][jtile(64)][ktile(32)][lane(32)][e(16)]
//   value = W[ktile*32 + (lane<16?0:8) + e + (e<8?0:8)][jtile*16 + (lane&15)]
// Xg fragment layout (bf16): [mtile(2048)][gate(4)][jt(64)][lane(32)][r(8)]
// s  fragment layout (f32):  [mtile(4)][jt(64)][lane(32)][r(8)]

__global__ __launch_bounds__(256) void lstm_pack_w(
    const float* __restrict__ W0, const float* __restrict__ W1,
    const float* __restrict__ W2, const float* __restrict__ W3,
    __bf16* __restrict__ out) {
  unsigned idx = blockIdx.x * 256u + threadIdx.x;           // 4*64*32*32*16 = 4194304
  if (idx >= 4u * 64u * 32u * 32u * 16u) return;
  int e    = idx & 15;
  int lane = (idx >> 4)  & 31;
  int kt   = (idx >> 9)  & 31;
  int jt   = (idx >> 14) & 63;
  int g    = (idx >> 20) & 3;
  const float* W = (g == 0) ? W0 : (g == 1) ? W1 : (g == 2) ? W2 : W3;
  int n = jt * 16 + (lane & 15);
  int k = kt * 32 + ((lane < 16) ? 0 : 8) + e + ((e < 8) ? 0 : 8);
  out[idx] = (__bf16)W[(size_t)k * H_DIM + n];
}

__global__ __launch_bounds__(256) void lstm_init_state(
    __bf16* __restrict__ h0, __bf16* __restrict__ h1, float* __restrict__ s) {
  unsigned i = blockIdx.x * 256u + threadIdx.x;             // 65536 threads
  if (i < (unsigned)(B_DIM * H_DIM)) {
    h0[i] = (__bf16)0.0f;
    h1[i] = (__bf16)0.0f;
    s[i]  = 0.0f;
  }
}

// ---------------------------------------------------------------------------
// Phase 1: Xg = X @ Wx + b, fragment-order bf16 output.
// grid = (16, 2048), block = 128 (4 waves; wave handles 4 consecutive N-tiles
// sharing one A fragment). One-iteration software pipeline on A and B.
// ---------------------------------------------------------------------------
__global__ __launch_bounds__(128) void lstm_xgemm(
    const float* __restrict__ X, const __bf16* __restrict__ pWx,
    const float* __restrict__ ba, const float* __restrict__ bi,
    const float* __restrict__ bf_, const float* __restrict__ bo,
    __bf16* __restrict__ Xg) {
  const int lane   = threadIdx.x & 31;
  const int wv     = threadIdx.x >> 5;
  const int ntbase = (blockIdx.x * 4 + wv) * 4;   // 4 N-tiles: ntbase..ntbase+3
  const int mtile  = blockIdx.y;                  // 0..2047 over T*B
  const int arow   = mtile * 16 + (lane & 15);    // A row (t*B+b)
  const int asel   = (lane < 16) ? 0 : 8;         // A K-phase per lane group

  const __bf16* bp = pWx + ((size_t)ntbase * 32) * 512 + lane * 16;
  const float*  xp = X + (size_t)arow * I_DIM + asel;

  auto loadA = [&](int kt) -> v16bf {
    const float* xr = xp + kt * 32;
    v4f f0 = *(const v4f*)(xr);
    v4f f1 = *(const v4f*)(xr + 4);
    v4f f2 = *(const v4f*)(xr + 16);
    v4f f3 = *(const v4f*)(xr + 20);
    v16bf a;
#pragma unroll
    for (int q = 0; q < 4; ++q) {
      a[q]      = (__bf16)f0[q];
      a[4 + q]  = (__bf16)f1[q];
      a[8 + q]  = (__bf16)f2[q];
      a[12 + q] = (__bf16)f3[q];
    }
    return a;
  };

  v8f acc[4];
#pragma unroll
  for (int u = 0; u < 4; ++u) acc[u] = (v8f){};

  // Pipeline prologue: data for kt = 0.
  v16bf acur = loadA(0);
  v16bf bcur[4];
#pragma unroll
  for (int u = 0; u < 4; ++u)
    bcur[u] = *(const v16bf*)(bp + (size_t)u * 32 * 512);

#pragma unroll 4
  for (int kt = 0; kt < 31; ++kt) {
    // Issue next iteration's loads before this iteration's WMMAs.
    v16bf anext = loadA(kt + 1);
    v16bf bnext[4];
#pragma unroll
    for (int u = 0; u < 4; ++u)
      bnext[u] = *(const v16bf*)(bp + (size_t)u * 32 * 512 + (size_t)(kt + 1) * 512);
#pragma unroll
    for (int u = 0; u < 4; ++u)
      acc[u] = __builtin_amdgcn_wmma_f32_16x16x32_bf16(false, acur, false, bcur[u],
                                                       (short)0, acc[u], false, false);
    acur = anext;
#pragma unroll
    for (int u = 0; u < 4; ++u) bcur[u] = bnext[u];
  }
#pragma unroll
  for (int u = 0; u < 4; ++u)
    acc[u] = __builtin_amdgcn_wmma_f32_16x16x32_bf16(false, acur, false, bcur[u],
                                                     (short)0, acc[u], false, false);

  const int gate = ntbase >> 6;                   // uniform across the 4 tiles
  const float* bptr = (gate == 0) ? ba : (gate == 1) ? bi : (gate == 2) ? bf_ : bo;
#pragma unroll
  for (int u = 0; u < 4; ++u) {
    const int jt  = (ntbase + u) & 63;
    const float bv = bptr[jt * 16 + (lane & 15)];
    v8bf o;
#pragma unroll
    for (int r = 0; r < 8; ++r) o[r] = (__bf16)(acc[u][r] + bv);
    // fragment-order store: [mtile][gate][jt][lane][r]
    *(v8bf*)(Xg + ((((size_t)mtile * 4 + gate) * 64 + jt) * 32 + lane) * 8) = o;
  }
}

// ---------------------------------------------------------------------------
// Phase 2: one recurrent step. grid = (64, 2), block = 32 (single wave).
// Each wave: 2 M-tiles (32 batch rows) x 16 hidden cols x 4 gates:
// 8 WMMA chains sharing 4 B fragments. One-iteration software pipeline.
// ---------------------------------------------------------------------------
__global__ __launch_bounds__(32) void lstm_step(
    const __bf16* __restrict__ hprev, __bf16* __restrict__ hnext,
    float* __restrict__ s, const __bf16* __restrict__ Xg_t,
    const __bf16* __restrict__ pWh, float* __restrict__ out_t) {
  const int lane  = threadIdx.x & 31;
  const int jt    = blockIdx.x;                 // 0..63
  const int mpair = blockIdx.y;                 // 0..1 -> mtiles 2*mpair, 2*mpair+1
  const int asel  = (lane < 16) ? 0 : 8;
  const int arow0 = (2 * mpair) * 16 + (lane & 15);

  const __bf16* bb  = pWh + ((size_t)jt * 32) * 512 + lane * 16;
  const __bf16* hp0 = hprev + (size_t)arow0 * H_DIM + asel;

  auto loadA = [&](int kt, int m) -> v16bf {
    const __bf16* hr = hp0 + (size_t)m * 16 * H_DIM + kt * 32;
    v8bf lo = *(const v8bf*)(hr);
    v8bf hi = *(const v8bf*)(hr + 16);
    v16bf a;
#pragma unroll
    for (int q = 0; q < 8; ++q) { a[q] = lo[q]; a[8 + q] = hi[q]; }
    return a;
  };

  v8f acc[2][4];
#pragma unroll
  for (int m = 0; m < 2; ++m)
#pragma unroll
    for (int g = 0; g < 4; ++g) acc[m][g] = (v8f){};

  // Pipeline prologue: data for kt = 0.
  v16bf acur[2] = { loadA(0, 0), loadA(0, 1) };
  v16bf wcur[4];
#pragma unroll
  for (int g = 0; g < 4; ++g)
    wcur[g] = *(const v16bf*)(bb + (size_t)g * 64 * 32 * 512);

#pragma unroll 4
  for (int kt = 0; kt < 31; ++kt) {
    v16bf anext[2] = { loadA(kt + 1, 0), loadA(kt + 1, 1) };
    v16bf wnext[4];
#pragma unroll
    for (int g = 0; g < 4; ++g)
      wnext[g] = *(const v16bf*)(bb + (size_t)g * 64 * 32 * 512 + (size_t)(kt + 1) * 512);
#pragma unroll
    for (int g = 0; g < 4; ++g) {
      acc[0][g] = __builtin_amdgcn_wmma_f32_16x16x32_bf16(false, acur[0], false, wcur[g],
                                                          (short)0, acc[0][g], false, false);
      acc[1][g] = __builtin_amdgcn_wmma_f32_16x16x32_bf16(false, acur[1], false, wcur[g],
                                                          (short)0, acc[1][g], false, false);
    }
    acur[0] = anext[0]; acur[1] = anext[1];
#pragma unroll
    for (int g = 0; g < 4; ++g) wcur[g] = wnext[g];
  }
#pragma unroll
  for (int g = 0; g < 4; ++g) {
    acc[0][g] = __builtin_amdgcn_wmma_f32_16x16x32_bf16(false, acur[0], false, wcur[g],
                                                        (short)0, acc[0][g], false, false);
    acc[1][g] = __builtin_amdgcn_wmma_f32_16x16x32_bf16(false, acur[1], false, wcur[g],
                                                        (short)0, acc[1][g], false, false);
  }

  const int colH = jt * 16 + (lane & 15);
#pragma unroll
  for (int m = 0; m < 2; ++m) {
    const int mtile = 2 * mpair + m;
    const int rbase = mtile * 16 + ((lane >> 4) << 3);
    // fragment-order gate addends (bf16) and cell state (f32)
    const size_t xgb = (((size_t)mtile * 4) * 64 + jt) * 32 * 8 + (size_t)lane * 8;
    v8bf xa = *(const v8bf*)(Xg_t + xgb + 0 * 64 * 32 * 8);
    v8bf xi = *(const v8bf*)(Xg_t + xgb + 1 * 64 * 32 * 8);
    v8bf xf = *(const v8bf*)(Xg_t + xgb + 2 * 64 * 32 * 8);
    v8bf xo = *(const v8bf*)(Xg_t + xgb + 3 * 64 * 32 * 8);
    float* sp = s + (((size_t)mtile * 64 + jt) * 32 + lane) * 8;
    v8f sv = *(const v8f*)sp;
    v8f sn;
#pragma unroll
    for (int r = 0; r < 8; ++r) {
      float ga = acc[m][0][r] + (float)xa[r];
      float gi = acc[m][1][r] + (float)xi[r];
      float gf = acc[m][2][r] + (float)xf[r];
      float go = acc[m][3][r] + (float)xo[r];
      // tanh(x) = (1-e^-2x) * rcp(1+e^-2x) ; sigmoid(x) = rcp(1+e^-x)
      float ea = __expf(-2.0f * ga);
      float av = (1.0f - ea) * __builtin_amdgcn_rcpf(1.0f + ea);
      float iv = __builtin_amdgcn_rcpf(1.0f + __expf(-gi));
      float fv = __builtin_amdgcn_rcpf(1.0f + __expf(-gf));
      float ov = __builtin_amdgcn_rcpf(1.0f + __expf(-go));
      sn[r] = av * iv + sv[r] * fv;
      float et = __expf(-2.0f * sn[r]);
      float hv = (1.0f - et) * __builtin_amdgcn_rcpf(1.0f + et) * ov;
      const int row = rbase + r;
      hnext[(size_t)row * H_DIM + colH] = (__bf16)hv;  // row-major for next A
      out_t[(size_t)row * H_DIM + colH] = hv;          // reference layout
    }
    *(v8f*)sp = sn;
  }
}

// ---------------------------------------------------------------------------
// Host launcher.  Inputs (dict order):
//  0:X 1:Wa_x 2:Wi_x 3:Wf_x 4:Wo_x 5:Wa_h 6:Wi_h 7:Wf_h 8:Wo_h 9:ba 10:bi 11:bf 12:bo
// Workspace: pWx 8MB | pWh 8MB | Xg 256MB | h0 128K | h1 128K | s 256K
// ---------------------------------------------------------------------------
extern "C" void kernel_launch(void* const* d_in, const int* in_sizes, int n_in,
                              void* d_out, int out_size, void* d_ws, size_t ws_size,
                              hipStream_t stream) {
  const float* X    = (const float*)d_in[0];
  const float* Wax  = (const float*)d_in[1];
  const float* Wix  = (const float*)d_in[2];
  const float* Wfx  = (const float*)d_in[3];
  const float* Wox  = (const float*)d_in[4];
  const float* Wah  = (const float*)d_in[5];
  const float* Wih  = (const float*)d_in[6];
  const float* Wfh  = (const float*)d_in[7];
  const float* Woh  = (const float*)d_in[8];
  const float* ba   = (const float*)d_in[9];
  const float* bi   = (const float*)d_in[10];
  const float* bf_  = (const float*)d_in[11];
  const float* bo   = (const float*)d_in[12];

  char* ws = (char*)d_ws;
  const size_t SZ_PW  = (size_t)4 * 64 * 32 * 32 * 16 * sizeof(__bf16); // 8 MB
  const size_t SZ_XG  = (size_t)T_DIM * B_DIM * G4H * sizeof(__bf16);   // 256 MB
  const size_t SZ_H   = (size_t)B_DIM * H_DIM * sizeof(__bf16);         // 128 KB
  __bf16* pWx = (__bf16*)(ws);
  __bf16* pWh = (__bf16*)(ws + SZ_PW);
  __bf16* Xg  = (__bf16*)(ws + 2 * SZ_PW);
  __bf16* h0  = (__bf16*)(ws + 2 * SZ_PW + SZ_XG);
  __bf16* h1  = (__bf16*)(ws + 2 * SZ_PW + SZ_XG + SZ_H);
  float*  s   = (float*) (ws + 2 * SZ_PW + SZ_XG + 2 * SZ_H);
  float*  out = (float*)d_out;

  // Pack weights into WMMA B-fragment order (bf16).
  lstm_pack_w<<<16384, 256, 0, stream>>>(Wax, Wix, Wfx, Wox, pWx);
  lstm_pack_w<<<16384, 256, 0, stream>>>(Wah, Wih, Wfh, Woh, pWh);
  lstm_init_state<<<256, 256, 0, stream>>>(h0, h1, s);

  // Phase 1: fused input projection across all timesteps.
  lstm_xgemm<<<dim3(16, (T_DIM * B_DIM) / 16), 128, 0, stream>>>(
      X, pWx, ba, bi, bf_, bo, Xg);

  // Phase 2: serialized recurrence; kernel boundary = grid-wide sync.
  __bf16* hp = h0;
  __bf16* hn = h1;
  for (int t = 0; t < T_DIM; ++t) {
    lstm_step<<<dim3(H_DIM / 16, 2), 32, 0, stream>>>(
        hp, hn, s, Xg + (size_t)t * B_DIM * G4H, pWh,
        out + (size_t)t * B_DIM * H_DIM);
    __bf16* tmp = hp; hp = hn; hn = tmp;
  }
}
